// Transformer2_72593537237175
// MI455X (gfx1250) — matrix operand order repaired and checked by reading, not compiled, not run on hardware
//
#include <hip/hip_runtime.h>

// ---------------- dims ----------------
constexpr int cB = 8, cN = 1024, cR = 128, cT = 9, cE = 64, cH = 2, cE2 = 128;

typedef __attribute__((ext_vector_type(16))) _Float16 v16h;
typedef __attribute__((ext_vector_type(8)))  _Float16 v8h;
typedef __attribute__((ext_vector_type(8)))  float    v8f;

// =====================================================================
// Weight conversion: WT[n*K + k] = (f16) W[k*N + n]   (one-time per use)
// =====================================================================
__global__ void k_cvt_wt(const float* __restrict__ W, _Float16* __restrict__ WT,
                         int K, int N)
{
  long i = (long)blockIdx.x * blockDim.x + threadIdx.x;
  if (i >= (long)K * N) return;
  int n = (int)(i % N);
  int k = (int)(i / N);
  WT[(long)n * K + k] = (_Float16)W[i];
}

// =====================================================================
// WMMA GEMM:  Y = act( A[M,K] @ W[K,N] + bias ), batched over z.
// A: f32 row-major (lda); WT: f16 transposed [N][K]; Y: f32 (ldy).
// Requirements (satisfied by all call sites): K % 32 == 0, A/WT 16B aligned.
// Branch-free inner loop: rows/cols clamped, garbage masked at the store.
// Block = 128 threads = 4 waves; each wave one 16x16 D tile -> 64x16 tile.
// act: 0=none 1=relu 2=sigmoid
// =====================================================================
__global__ __launch_bounds__(128) void k_gemm(const float* __restrict__ A,
                                              const _Float16* __restrict__ WT,
                                              const float* __restrict__ bias,
                                              float* __restrict__ Y,
                                              int M, int N, int K,
                                              int lda, int ldy,
                                              long sA, long sY, int act)
{
  const int wid  = threadIdx.x >> 5;
  const int lane = threadIdx.x & 31;
  const int n0 = blockIdx.x * 16;
  const int m0 = blockIdx.y * 64;

  const int row  = m0 + wid * 16 + (lane & 15);
  const int rowc = (row < M) ? row : (M - 1);          // clamp: always valid
  const int col  = n0 + (lane & 15);
  const int colc = (col < N) ? col : (N - 1);

  // A: per-lane row, two 8-float runs at k0 and k0+16 (16-bit A layout).
  const float* ap = A + (long)blockIdx.z * sA + (long)rowc * lda + ((lane < 16) ? 0 : 8);
  // B: per-lane column of W == row of WT, two 8-half runs at k0 and k0+16 (16-bit B layout).
  const _Float16* bp = WT + (long)colc * K + ((lane < 16) ? 0 : 8);

  v8f acc = {0.f, 0.f, 0.f, 0.f, 0.f, 0.f, 0.f, 0.f};
  for (int k0 = 0; k0 < K; k0 += 32) {
    const float4 a0 = *(const float4*)(ap + k0);
    const float4 a1 = *(const float4*)(ap + k0 + 4);
    const float4 a2 = *(const float4*)(ap + k0 + 16);
    const float4 a3 = *(const float4*)(ap + k0 + 20);
    const v8h b0 = *(const v8h*)(bp + k0);
    const v8h b1 = *(const v8h*)(bp + k0 + 16);
    v16h af, bf;
    af[0]  = (_Float16)a0.x; af[1]  = (_Float16)a0.y;
    af[2]  = (_Float16)a0.z; af[3]  = (_Float16)a0.w;
    af[4]  = (_Float16)a1.x; af[5]  = (_Float16)a1.y;
    af[6]  = (_Float16)a1.z; af[7]  = (_Float16)a1.w;
    af[8]  = (_Float16)a2.x; af[9]  = (_Float16)a2.y;
    af[10] = (_Float16)a2.z; af[11] = (_Float16)a2.w;
    af[12] = (_Float16)a3.x; af[13] = (_Float16)a3.y;
    af[14] = (_Float16)a3.z; af[15] = (_Float16)a3.w;
    #pragma unroll
    for (int i = 0; i < 8; ++i) { bf[i] = b0[i]; bf[8 + i] = b1[i]; }
    acc = __builtin_amdgcn_wmma_f32_16x16x32_f16(false, af, false, bf,
                                                 (short)0, acc, false, false);
  }

  if (col < N) {
    float* yp = Y + (long)blockIdx.z * sY;
    const float bv = bias ? bias[col] : 0.f;
    #pragma unroll
    for (int r = 0; r < 8; ++r) {
      const int orow = m0 + wid * 16 + ((lane < 16) ? r : r + 8);
      if (orow < M) {
        float v = acc[r] + bv;
        if (act == 1)      v = fmaxf(v, 0.f);
        else if (act == 2) v = 1.f / (1.f + __expf(-v));
        yp[(long)orow * ldy + col] = v;
      }
    }
  }
}

// ---------------- q2 = src + ds[n] broadcast ----------------
__global__ void k_add_ds(const float* __restrict__ src, const float* __restrict__ ds,
                         float* __restrict__ out, long total)
{
  long i = (long)blockIdx.x * blockDim.x + threadIdx.x;
  if (i >= total) return;
  int e = (int)(i & 63);
  int n = (int)((i / (cT * cE)) % cN);
  out[i] = src[i] + ds[n * cE + e];
}

// ---------------- spatial attention scores ----------------
// S[b,q,k,t,h] = clip( (qh.kh * m + NEG*(1-m)) * scale, -5, 5 )
// qmode 1: Q = Qr[q, E] broadcast over (b,t); qmode 0: Q is (B,Qn,T,E)
// maskmode 1: m = adj[k*R+q] (adj_sr.T) ; 2: m = adj[q*R+k] (adj_r) ; 0: none
__global__ void k_scores(float* __restrict__ S, const float* __restrict__ Q,
                         const float* __restrict__ Kh, const int* __restrict__ adj,
                         int Qn, int Kn, int qmode, int maskmode, float scale)
{
  long idx = (long)blockIdx.x * blockDim.x + threadIdx.x;
  long total = (long)cB * Qn * Kn * cT * cH;
  if (idx >= total) return;
  long r = idx;
  int h = (int)(r % cH); r /= cH;
  int t = (int)(r % cT); r /= cT;
  int k = (int)(r % Kn); r /= Kn;
  int q = (int)(r % Qn);
  int b = (int)(r / Qn);
  const float* qp = (qmode == 1) ? (Q + (long)q * cE + h * 32)
                                 : (Q + ((((long)b * Qn + q) * cT + t) * cE + h * 32));
  const float* kp = Kh + ((((long)b * Kn + k) * cT + t) * cE + h * 32);
  float s = 0.f;
  #pragma unroll
  for (int j = 0; j < 32; ++j) s += qp[j] * kp[j];
  if (maskmode == 1)      { float m = (float)adj[(long)k * cR + q]; s = s * m + (-1e8f) * (1.f - m); }
  else if (maskmode == 2) { float m = (float)adj[q * cR + k];       s = s * m + (-1e8f) * (1.f - m); }
  s *= scale;
  s = fminf(5.f, fmaxf(-5.f, s));
  S[idx] = s;
}

// softmax over the q axis of S[b,q,k,t,h]; one wave32 per (b,k,t,h) column.
// Scores pre-clipped to [-5,5] -> exp is safe without a max pass.
__global__ __launch_bounds__(128) void k_softmax_q(float* __restrict__ S, int Qn, int Kn)
{
  long ncol = (long)cB * Kn * cT * cH;
  long col  = (long)blockIdx.x * 4 + (threadIdx.x >> 5);
  int lane  = threadIdx.x & 31;
  if (col >= ncol) return;
  long r = col;
  int h = (int)(r % cH); r /= cH;
  int t = (int)(r % cT); r /= cT;
  int k = (int)(r % Kn);
  int b = (int)(r / Kn);
  long base = (((long)b * Qn * Kn + k) * cT + t) * cH + h;
  long qs   = (long)Kn * cT * cH;
  float sum = 0.f;
  for (int q = lane; q < Qn; q += 32) sum += __expf(S[base + (long)q * qs]);
  #pragma unroll
  for (int off = 16; off > 0; off >>= 1) sum += __shfl_xor(sum, off, 32);
  float inv = 1.f / sum;
  for (int q = lane; q < Qn; q += 32) {
    long p = base + (long)q * qs;
    S[p] = __expf(S[p]) * inv;
  }
}

// O[b,q,t,e] = sum_k S[b,q,k,t,h(e)] * Vh[b,k,t,e]
__global__ void k_av(float* __restrict__ O, const float* __restrict__ S,
                     const float* __restrict__ Vh, int Qn, int Kn)
{
  long idx = (long)blockIdx.x * blockDim.x + threadIdx.x;
  long total = (long)cB * Qn * cT * cE;
  if (idx >= total) return;
  long r = idx;
  int e = (int)(r % cE); r /= cE;
  int t = (int)(r % cT); r /= cT;
  int q = (int)(r % Qn);
  int b = (int)(r / Qn);
  int h = e >> 5;
  long sbase = ((((long)b * Qn + q) * Kn) * cT + t) * cH + h;
  const float* vp = Vh + ((long)b * Kn * cT + t) * cE + e;
  float acc = 0.f;
  for (int k = 0; k < Kn; ++k)
    acc += S[sbase + (long)k * (cT * cH)] * vp[(long)k * (cT * cE)];
  O[idx] = acc;
}

// ---------------- temporal attention (C=128, H=2, d=64), one wave per (b,n) ----------------
// Single-wave workgroup: barriers degrade to S_NOP per ISA; scores in LDS.
__global__ __launch_bounds__(32) void k_tattn(float* __restrict__ O, const float* __restrict__ Qh,
                                              const float* __restrict__ Kh, const float* __restrict__ Vh,
                                              int Tq, int Tk, int causal, float scale)
{
  __shared__ float sc[2 * 9 * 9];
  const int C = cE2, d = 64;
  int lane = threadIdx.x & 31;
  long bn = blockIdx.x;
  const float* qb = Qh + bn * Tq * C;
  const float* kb = Kh + bn * Tk * C;
  const float* vb = Vh + bn * Tk * C;
  float* ob = O + bn * Tq * C;

  int nsc = cH * Tq * Tk;
  for (int i = lane; i < nsc; i += 32) {
    int h = i / (Tq * Tk);
    int rem = i - h * Tq * Tk;
    int q = rem / Tk, k = rem - (rem / Tk) * Tk;
    float s = 0.f;
    const float* qp = qb + q * C + h * d;
    const float* kp = kb + k * C + h * d;
    #pragma unroll 16
    for (int j = 0; j < d; ++j) s += qp[j] * kp[j];
    if (causal && k > q) s = -1e8f;
    s *= scale;
    s = fminf(5.f, fmaxf(-5.f, s));
    sc[i] = __expf(s);
  }
  __syncthreads();
  for (int i = lane; i < cH * Tk; i += 32) {     // softmax over q per (h,k)
    int h = i / Tk, k = i - (i / Tk) * Tk;
    float sum = 0.f;
    for (int q = 0; q < Tq; ++q) sum += sc[(h * Tq + q) * Tk + k];
    float inv = 1.f / sum;
    for (int q = 0; q < Tq; ++q) sc[(h * Tq + q) * Tk + k] *= inv;
  }
  __syncthreads();
  for (int i = lane; i < Tq * C; i += 32) {
    int q = i / C, e = i - (i / C) * C;
    int h = e >> 6;
    float acc = 0.f;
    for (int k = 0; k < Tk; ++k)
      acc += sc[(h * Tq + q) * Tk + k] * vb[k * C + e];
    ob[i] = acc;
  }
}

// ---------------- layernorm: Y = LN(X (+Res)) * g + b; one wave32 per row ----------------
__global__ __launch_bounds__(128) void k_ln(float* __restrict__ Y, const float* __restrict__ X,
                                            const float* __restrict__ Res,
                                            const float* __restrict__ g, const float* __restrict__ bb,
                                            long Mrows, int C)
{
  long row = (long)blockIdx.x * 4 + (threadIdx.x >> 5);
  int lane = threadIdx.x & 31;
  if (row >= Mrows) return;
  const float* xp = X + row * C;
  const float* rp = Res ? Res + row * C : nullptr;
  int per = C >> 5;           // 2 (C=64) or 4 (C=128)
  float v[4];
  float sum = 0.f;
  for (int j = 0; j < per; ++j) {
    float t = xp[lane + j * 32];
    if (rp) t += rp[lane + j * 32];
    v[j] = t; sum += t;
  }
  #pragma unroll
  for (int off = 16; off > 0; off >>= 1) sum += __shfl_xor(sum, off, 32);
  float mean = sum / (float)C;
  float var = 0.f;
  for (int j = 0; j < per; ++j) { float d = v[j] - mean; var += d * d; }
  #pragma unroll
  for (int off = 16; off > 0; off >>= 1) var += __shfl_xor(var, off, 32);
  var /= (float)C;
  float rs = rsqrtf(var + 1e-5f);
  float* yp = Y + row * C;
  for (int j = 0; j < per; ++j) {
    int c = lane + j * 32;
    yp[c] = (v[j] - mean) * rs * g[c] + bb[c];
  }
}

// ---------------- elementwise helpers ----------------
__global__ void k_concat_pad(float* __restrict__ out, const float* __restrict__ a, long total)
{
  long i = (long)blockIdx.x * blockDim.x + threadIdx.x;
  if (i >= total) return;
  int e = (int)(i & 127); long row = i >> 7;
  out[i] = (e < cE) ? a[row * cE + e] : 0.f;
}
__global__ void k_concat2(float* __restrict__ out, const float* __restrict__ a,
                          const float* __restrict__ b, long total)
{
  long i = (long)blockIdx.x * blockDim.x + threadIdx.x;
  if (i >= total) return;
  int e = (int)(i & 127); long row = i >> 7;
  out[i] = (e < cE) ? a[row * cE + e] : b[row * cE + (e - cE)];
}
__global__ void k_tile_tgt1(float* __restrict__ out, const float* __restrict__ senc, long total)
{
  long i = (long)blockIdx.x * blockDim.x + threadIdx.x;
  if (i >= total) return;
  int e = (int)(i & 127); long r = i >> 7;
  int q = (int)(r % 3); long bn = r / 3;
  out[i] = senc[((bn * cT) + 6 + q) * cE + (e & 63)];
}
__global__ void k_absdiff(float* __restrict__ out, const float* __restrict__ senc,
                          const float* __restrict__ x1, long total)
{
  long i = (long)blockIdx.x * blockDim.x + threadIdx.x;
  if (i >= total) return;
  int e = (int)(i & 63); long r = i >> 6;
  int t = (int)(r % cT); long bn = r / cT;
  out[i] = fabsf(senc[i] - x1[((bn * 3) + (t % 3)) * cE + e]);
}

// =====================================================================
// Host orchestration.
// Input index map (setup_inputs dict insertion order, params flattened
// recursively in insertion order):
//  0 src | st: 1 D_S 2 We 3 be 4 I 5-9 sr(Wv,Wk,Wq,Wo,bo) 10-14 rr 15-19 rs
//  20 g1 21 b1 22 g2 23 b2 24 W1 25 bf1 26 W2 27 bf2 28 Wfs 29 bfs
//  30 g1 31 b1 32 g2 33 b2 | ttf: 34-38 att 39 g1 40 b1 41 g2 42 b2
//  43 W1 44 bf1 45 W2 46 bf2 | dec1: 47-59 tt1, 60-72 tt2, 73 g1 74 b1 75 g2 76 b2
//  dec2: 77-106 (same layout) | 107 Wfcn 108 bfcn 109 Wln 110 bln
//  111 adj_sr 112 adj_r 113 t
// Workspace: ~95M floats (~380 MB) + 128KB f16 weight scratch.
// =====================================================================
extern "C" void kernel_launch(void* const* d_in, const int* in_sizes, int n_in,
                              void* d_out, int out_size, void* d_ws, size_t ws_size,
                              hipStream_t stream)
{
  (void)in_sizes; (void)n_in; (void)out_size; (void)ws_size;
  auto F = [&](int i) { return (const float*)d_in[i]; };
  const int* adj_sr = (const int*)d_in[111];
  const int* adj_r  = (const int*)d_in[112];
  float* out = (float*)d_out;
  float* ws  = (float*)d_ws;

  const long SZ_E   = (long)cB * cN * cT * cE;    // 4718592
  const long SZ_E2  = (long)cB * cN * cT * cE2;   // 9437184
  const long SZ_R   = (long)cB * cR * cT * cE;    // 589824
  const long SZ_3E  = (long)cB * cN * 3 * cE;     // 1572864
  const long SZ_3E2 = (long)cB * cN * 3 * cE2;    // 3145728

  // workspace layout
  float* q2    = ws;
  float* tA    = q2 + SZ_E;
  float* tB    = tA + SZ_E;
  float* tC    = tB + SZ_E;
  float* tD    = tC + SZ_E;
  float* senc  = tD + SZ_E;
  float* g0    = senc + SZ_E;
  float* g1b   = g0 + SZ_E2;
  float* g2b   = g1b + SZ_E2;
  float* g3b   = g2b + SZ_E2;
  float* g4b   = g3b + SZ_E2;
  float* hid   = g4b + SZ_E2;
  float* dsb   = hid + SZ_E2;
  float* qhr   = dsb + (long)cN * cE;
  float* rA    = qhr + (long)cR * cE;
  float* rB    = rA + SZ_R;
  float* rC    = rB + SZ_R;
  float* rD    = rC + SZ_R;
  float* rE    = rD + SZ_R;
  float* rF    = rE + SZ_R;
  float* x1buf = rF + SZ_R;
  float* x2buf = x1buf + SZ_3E;
  float* tgt1b = x2buf + SZ_3E;
  _Float16* wtbuf = (_Float16*)(tgt1b + SZ_3E2);  // 65536 halves (max K*N)

  // output layout (tuple concat): x, c, o1, o2, asr, arr, ars, h
  float* OX   = out;
  float* OC   = OX + SZ_E2;
  float* OO1  = OC + SZ_E;
  float* OO2  = OO1 + (long)cB * cN * 3 * 2;
  float* OASR = OO2 + (long)cB * cN * 3 * 2;
  float* OARR = OASR + (long)cB * cR * cN * cT * cH;
  float* OARS = OARR + (long)cB * cR * cR * cT * cH;
  float* OH   = OARS + (long)cB * cN * cR * cT * cH;

  auto ew = [](long total) { return dim3((unsigned)((total + 255) / 256)); };
  // gemm: converts weight to f16-transposed scratch (stream-ordered reuse is
  // safe: kernels on one stream serialize), then runs the WMMA GEMM.
  auto gemm = [&](const float* A, const float* Wf, const float* bias, float* Y,
                  int M, int N, int K, int lda, int ldy,
                  long sA, long sY, int nb, int act) {
    k_cvt_wt<<<ew((long)K * N), 256, 0, stream>>>(Wf, wtbuf, K, N);
    dim3 g((N + 15) / 16, (M + 63) / 64, nb);
    k_gemm<<<g, 128, 0, stream>>>(A, wtbuf, bias, Y, M, N, K, lda, ldy, sA, sY, act);
  };
  auto headproj = [&](const float* X, const float* W, float* Y, int Mtok) {  // C=64, d=32
    gemm(X, W, nullptr, Y, Mtok, 32, 32, cE, cE, 32, 32, cH, 0);
  };
  auto headproj2 = [&](const float* X, const float* W, float* Y, int Mtok) { // C=128, d=64
    gemm(X, W, nullptr, Y, Mtok, 64, 64, cE2, cE2, 64, 64, cH, 0);
  };
  auto ln = [&](float* Y, const float* X, const float* Res, const float* g,
                const float* b, long Mrows, int C) {
    k_ln<<<dim3((unsigned)((Mrows + 3) / 4)), 128, 0, stream>>>(Y, X, Res, g, b, Mrows, C);
  };
  auto ffn = [&](const float* X, float* Yout, float* hidb, int Mtok, int Cw,
                 const float* W1, const float* b1, const float* W2, const float* b2,
                 int hidden, int chunk) {
    for (int m0 = 0; m0 < Mtok; m0 += chunk) {
      int mc = (Mtok - m0 < chunk) ? (Mtok - m0) : chunk;
      gemm(X + (long)m0 * Cw, W1, b1, hidb, mc, hidden, Cw, Cw, hidden, 0, 0, 1, 1);
      gemm(hidb, W2, b2, Yout + (long)m0 * Cw, mc, Cw, hidden, hidden, Cw, 0, 0, 1, 0);
    }
  };
  // temporal transformer block at C=128. s1: qh->WoOut->ffnOut, s2: kh->ln1,
  // s3: vh (may alias hidb), s4: attn o, hidb: ffn hidden scratch.
  auto ttrans128 = [&](const float* qin, const float* kin, const float* vin,
                       int Tq, int Tk, int causal, int base, float* outb,
                       float* s1, float* s2, float* s3, float* s4, float* hidb) {
    int Mq = 8192 * Tq, Mk = 8192 * Tk;
    headproj2(qin, F(base + 2), s1, Mq);
    headproj2(kin, F(base + 1), s2, Mk);
    headproj2(vin, F(base + 0), s3, Mk);
    k_tattn<<<dim3(8192), 32, 0, stream>>>(s4, s1, s2, s3, Tq, Tk, causal,
                                           0.08838834764831843f /* 1/sqrt(128) */);
    gemm(s4, F(base + 3), F(base + 4), s1, Mq, cE2, cE2, cE2, cE2, 0, 0, 1, 0);
    ln(s2, s1, qin, F(base + 5), F(base + 6), Mq, cE2);
    ffn(s2, s1, hidb, Mq, cE2, F(base + 9), F(base + 10), F(base + 11), F(base + 12), 512, 18432);
    ln(outb, s1, s2, F(base + 7), F(base + 8), Mq, cE2);
  };
  auto decoder = [&](const float* xin, int dbase, float* dout) {
    float* c0 = g2b; float* c1 = g2b + SZ_3E2; float* c2 = g2b + 2 * SZ_3E2;
    float* d0 = g3b; float* d1 = g3b + SZ_3E2;
    ttrans128(tgt1b, tgt1b, tgt1b, 3, 3, 1, dbase, d0, c0, c1, c2, g4b, hid);
    ln(d1, d0, tgt1b, F(dbase + 26), F(dbase + 27), 24576, cE2);
    ttrans128(d1, xin, xin, 3, 9, 0, dbase + 13, c2, c0, g4b, hid, c1, hid);
    ln(dout, c2, d1, F(dbase + 28), F(dbase + 29), 24576, cE2);
  };

  // ===================== spatial transformer (_strans) =====================
  gemm(F(1), F(2), F(3), dsb, cN, cE, cR, cR, cE, 0, 0, 1, 0);               // ds = D_S@We+be
  k_add_ds<<<ew(SZ_E), 256, 0, stream>>>(F(0), dsb, q2, SZ_E);               // q2=k2=src+ds
  headproj(F(0), F(5), tA, 73728);                                           // vh (from src)
  headproj(q2,  F(6), tB, 73728);                                            // kh (from k2)
  headproj(F(4), F(7), qhr, cR);                                             // qh from I
  k_scores<<<ew((long)cB*cR*cN*cT*cH), 256, 0, stream>>>(OASR, qhr, tB, adj_sr, cR, cN, 1, 1, 0.125f);
  k_softmax_q<<<dim3((unsigned)(((long)cB*cN*cT*cH + 3) / 4)), 128, 0, stream>>>(OASR, cR, cN);
  k_av<<<ew((long)cB*cR*cT*cE), 256, 0, stream>>>(rA, OASR, tA, cR, cN);
  gemm(rA, F(8), F(9), rB, 9216, cE, cE, cE, cE, 0, 0, 1, 0);                // h1 = o@Wo+bo
  headproj(rB, F(10), rC, 9216);                                             // rr: vh
  headproj(rB, F(11), rD, 9216);                                             // rr: kh
  headproj(rB, F(12), rE, 9216);                                             // rr: qh
  k_scores<<<ew((long)cB*cR*cR*cT*cH), 256, 0, stream>>>(OARR, rE, rD, adj_r, cR, cR, 0, 2, 0.125f);
  k_softmax_q<<<dim3((unsigned)(((long)cB*cR*cT*cH + 3) / 4)), 128, 0, stream>>>(OARR, cR, cR);
  k_av<<<ew((long)cB*cR*cT*cE), 256, 0, stream>>>(rF, OARR, rC, cR, cR);
  gemm(rF, F(13), F(14), OH, 9216, cE, cE, cE, cE, 0, 0, 1, 0);              // h2 (output 'h')
  headproj(OH, F(15), rC, 9216);                                             // rs: vh
  headproj(OH, F(16), rD, 9216);                                             // rs: kh
  headproj(q2, F(17), tA, 73728);                                            // rs: qh (from q2)
  k_scores<<<ew((long)cB*cN*cR*cT*cH), 256, 0, stream>>>(OARS, tA, rD, nullptr, cN, cR, 0, 0, 0.125f);
  k_softmax_q<<<dim3((unsigned)(((long)cB*cR*cT*cH + 3) / 4)), 128, 0, stream>>>(OARS, cN, cR);
  k_av<<<ew(SZ_E), 256, 0, stream>>>(tB, OARS, rC, cN, cR);
  gemm(tB, F(18), F(19), tC, 73728, cE, cE, cE, cE, 0, 0, 1, 0);             // att
  ln(tD, tC, q2, F(20), F(21), 73728, cE);                                   // x = LN(att+q2)
  ffn(tD, tC, hid, 73728, cE, F(24), F(25), F(26), F(27), 256, 36864);       // ffn(x)
  ln(tA, tC, tD, F(22), F(23), 73728, cE);                                   // u
  gemm(tA, F(28), F(29), tC, 73728, cE, cE, cE, cE, 0, 0, 1, 0);             // x0 = u@Wfs+bfs

  // ===================== encoder / first branch =====================
  ln(senc, tC, F(0), F(30), F(31), 73728, cE);                               // senc
  k_concat_pad<<<ew(SZ_E2), 256, 0, stream>>>(g0, senc, SZ_E2);              // src1=[senc,0]
  ttrans128(g0, g0, g0, 9, 9, 0, 34, g1b, g2b, g3b, hid, g4b, hid);          // ttf(src1)
  ln(g0, g1b, g0, F(32), F(33), 73728, cE2);                                 // x = LN(.+src1)
  k_tile_tgt1<<<ew(SZ_3E2), 256, 0, stream>>>(tgt1b, senc, SZ_3E2);          // tgt1

  float* d2 = g3b + 2 * SZ_3E2;
  decoder(g0, 47, d2);                                                       // dec1
  gemm(d2, F(107), F(108), x1buf, 24576, cE, cE2, cE2, cE, 0, 0, 1, 2);      // x1=sigmoid
  gemm(x1buf, F(109), F(110), OO1, 24576, 2, cE, cE, 2, 0, 0, 1, 0);         // o1

  // ===================== second branch =====================
  k_absdiff<<<ew(SZ_E), 256, 0, stream>>>(OC, senc, x1buf, SZ_E);            // c
  k_concat2<<<ew(SZ_E2), 256, 0, stream>>>(g0, senc, OC, SZ_E2);             // src2=[senc,c]
  ttrans128(g0, g0, g0, 9, 9, 0, 34, g1b, g2b, g3b, hid, g4b, hid);          // ttf(src2)
  ln(OX, g1b, g0, F(32), F(33), 73728, cE2);                                 // x (output)
  decoder(OX, 77, d2);                                                       // dec2
  gemm(d2, F(107), F(108), x2buf, 24576, cE, cE2, cE2, cE, 0, 0, 1, 2);      // x2=sigmoid
  gemm(x2buf, F(109), F(110), OO2, 24576, 2, cE, cE, 2, 0, 0, 1, 0);         // o2
}
